// BitfieldLinear_70772471103880
// MI455X (gfx1250) — matrix operand order repaired and checked by reading, not compiled, not run on hardware
//
#include <hip/hip_runtime.h>

typedef __attribute__((ext_vector_type(16))) __bf16 v16bf;
typedef __attribute__((ext_vector_type(8)))  __bf16 v8bf;
typedef __attribute__((ext_vector_type(8)))  float  v8f;

// ---------------------------------------------------------------------------
// Kernel 1: decode W[o,i] = r(o) * basis[idx(o), i] + s(o) * (q[o,i]-128)/127
// into bf16 row-major [N, K].  One thread handles 8 consecutive i.
// ---------------------------------------------------------------------------
__global__ __launch_bounds__(256) void decode_w_kernel(
    const int*   __restrict__ codes,
    const float* __restrict__ basis,
    const int*   __restrict__ rq,
    const float* __restrict__ scales,
    __bf16*      __restrict__ Wb,
    int K)
{
    long long t  = (long long)blockIdx.x * blockDim.x + threadIdx.x;
    long long e0 = t * 8;
    int o  = (int)(e0 / K);
    int i0 = (int)(e0 - (long long)o * K);

    int   code = codes[o];
    int   bi   = code & 0xFF;
    float r    = (float)((code >> 8) & 0xFFFF) * (1.0f / 65535.0f); // * R_MAX(=1)
    float s    = scales[o] * (1.0f / 127.0f);

    const float4* bp = (const float4*)(basis + (long long)bi * K + i0);
    const int4*   qp = (const int4*)(rq + (long long)o * K + i0);
    float4 b0 = bp[0], b1 = bp[1];
    int4   q0 = qp[0], q1 = qp[1];

    v8bf w;
    w[0] = (__bf16)(r * b0.x + s * ((float)q0.x - 128.0f));
    w[1] = (__bf16)(r * b0.y + s * ((float)q0.y - 128.0f));
    w[2] = (__bf16)(r * b0.z + s * ((float)q0.z - 128.0f));
    w[3] = (__bf16)(r * b0.w + s * ((float)q0.w - 128.0f));
    w[4] = (__bf16)(r * b1.x + s * ((float)q1.x - 128.0f));
    w[5] = (__bf16)(r * b1.y + s * ((float)q1.y - 128.0f));
    w[6] = (__bf16)(r * b1.z + s * ((float)q1.z - 128.0f));
    w[7] = (__bf16)(r * b1.w + s * ((float)q1.w - 128.0f));
    *(v8bf*)(Wb + (long long)o * K + i0) = w;
}

// ---------------------------------------------------------------------------
// Kernel 1b: x fp32 -> bf16 (one streaming pass; amortized over N/BN reuses)
// ---------------------------------------------------------------------------
__global__ __launch_bounds__(256) void cvt_x_kernel(
    const float* __restrict__ X, __bf16* __restrict__ Xb)
{
    long long t = ((long long)blockIdx.x * blockDim.x + threadIdx.x) * 8;
    float4 f0 = *(const float4*)(X + t);
    float4 f1 = *(const float4*)(X + t + 4);
    v8bf h;
    h[0] = (__bf16)f0.x; h[1] = (__bf16)f0.y;
    h[2] = (__bf16)f0.z; h[3] = (__bf16)f0.w;
    h[4] = (__bf16)f1.x; h[5] = (__bf16)f1.y;
    h[6] = (__bf16)f1.z; h[7] = (__bf16)f1.w;
    *(v8bf*)(Xb + t) = h;
}

// ---------------------------------------------------------------------------
// Kernel 2: C[M,N] = X[M,K] @ Wb[N,K]^T + bias
// 128x128 block tile, BK=32, 8 wave32 waves (4x2), each wave 32x64 via
// 2x4 tiles of v_wmma_f32_16x16x32_bf16.  Double-buffered LDS.
// ABF16=true : both tiles staged with GLOBAL_LOAD_ASYNC_TO_LDS_B128
//              (no staging VALU in the main loop; tracked on ASYNCcnt).
// ABF16=false: A staged manually with fp32->bf16 cvt (small-workspace path).
// B fragments are software-pipelined (b_cur/b_next rotation) so the LDS
// loads for tile j+1 issue before the WMMAs of tile j consume b_cur.
// ---------------------------------------------------------------------------
#define BM 128
#define BN 128
#define BK 32
#define SAS 40   // padded LDS row stride in elements (80B -> conflict-free b128)

template <bool ABF16>
__global__ __launch_bounds__(256) void gemm_bf16_kernel(
    const float*  __restrict__ Xf,    // [M, K] fp32   (used when !ABF16)
    const __bf16* __restrict__ Xb,    // [M, K] bf16   (used when  ABF16)
    const __bf16* __restrict__ Wb,    // [N, K] bf16 (row n = column n of B)
    const float*  __restrict__ bias,  // [N]
    float*        __restrict__ C,     // [M, N]
    int M, int N, int K)
{
    __shared__ alignas(16) __bf16 sA[2][BM * SAS];
    __shared__ alignas(16) __bf16 sB[2][BN * SAS];

    const int tid  = threadIdx.x;
    const int bm   = blockIdx.y * BM;
    const int bn   = blockIdx.x * BN;

    const int wid  = tid >> 5;
    const int lane = tid & 31;
    const int wm   = wid & 3;        // 4 wave rows  -> 32 M-rows each
    const int wn   = wid >> 2;       // 2 wave cols  -> 64 N-cols each
    const int lr   = lane & 15;
    const int lh   = lane >> 4;

    v8f acc[2][4] = {};

    const int nk = K / BK;

    // ---- stage A ---------------------------------------------------------
    auto stageA = [&](int kt, int buf) {
        const int kcol = kt * BK;
#pragma unroll
        for (int it = 0; it < 2; ++it) {
            int c   = tid + it * 256;     // 0..511
            int row = c >> 2;             // 0..127
            int col = (c & 3) * 8;        // 0,8,16,24
            if (ABF16) {
                unsigned lds = (unsigned)(uintptr_t)(&sA[buf][row * SAS + col]);
                unsigned long long ga =
                    (unsigned long long)(const void*)(Xb + (long long)(bm + row) * K + kcol + col);
                asm volatile("global_load_async_to_lds_b128 %0, %1, off"
                             :: "v"(lds), "v"(ga) : "memory");
            } else {
                const float* ga = Xf + (long long)(bm + row) * K + kcol + col;
                float4 f0 = *(const float4*)ga;
                float4 f1 = *(const float4*)(ga + 4);
                v8bf ha;
                ha[0] = (__bf16)f0.x; ha[1] = (__bf16)f0.y;
                ha[2] = (__bf16)f0.z; ha[3] = (__bf16)f0.w;
                ha[4] = (__bf16)f1.x; ha[5] = (__bf16)f1.y;
                ha[6] = (__bf16)f1.z; ha[7] = (__bf16)f1.w;
                *(v8bf*)(&sA[buf][row * SAS + col]) = ha;
            }
        }
    };

    // ---- stage B via CDNA5 async global->LDS (tracked on ASYNCcnt) -------
    auto stageB = [&](int kt, int buf) {
        const int kcol = kt * BK;
#pragma unroll
        for (int it = 0; it < 2; ++it) {
            int c   = tid + it * 256;     // 0..511
            int row = c >> 2;             // 0..127
            int col = (c & 3) * 8;        // 0,8,16,24
            unsigned lds = (unsigned)(uintptr_t)(&sB[buf][row * SAS + col]);
            unsigned long long gaddr =
                (unsigned long long)(const void*)(Wb + (long long)(bn + row) * K + kcol + col);
            asm volatile("global_load_async_to_lds_b128 %0, %1, off"
                         :: "v"(lds), "v"(gaddr) : "memory");
        }
    };

    // ---- fragment loads (per CDNA5 16-bit WMMA VGPR layouts) -------------
    // A 16x32: lanes 0-15 row M=lr with K 0-7 (v0-3) & 16-23 (v4-7);
    //          lanes 16-31 same rows with K 8-15 & 24-31.
    auto loadA = [&](int buf, int tm) -> v16bf {
        const __bf16* p = &sA[buf][(tm + lr) * SAS + lh * 8];
        v8bf lo = *(const v8bf*)p;
        v8bf hi = *(const v8bf*)(p + 16);
        v16bf f;
#pragma unroll
        for (int j = 0; j < 8; ++j) { f[j] = lo[j]; f[j + 8] = hi[j]; }
        return f;
    };
    // B 32x16: lane = column N (W row); lanes 0-15 hold K 0-7 & 16-23,
    //          lanes 16-31 hold K 8-15 & 24-31.
    auto loadB = [&](int buf, int tn) -> v16bf {
        const __bf16* p = &sB[buf][(tn + lr) * SAS + lh * 8];
        v8bf lo = *(const v8bf*)p;
        v8bf hi = *(const v8bf*)(p + 16);
        v16bf f;
#pragma unroll
        for (int j = 0; j < 8; ++j) { f[j] = lo[j]; f[j + 8] = hi[j]; }
        return f;
    };

    stageB(0, 0);
    stageA(0, 0);
    asm volatile("s_wait_asynccnt 0x0" ::: "memory");
    __syncthreads();

    for (int kt = 0; kt < nk; ++kt) {
        const int buf = kt & 1;
        if (kt + 1 < nk) {
            stageB(kt + 1, buf ^ 1);   // async, overlaps with WMMA below
            stageA(kt + 1, buf ^ 1);
        }

        v16bf a0 = loadA(buf, wm * 32);
        v16bf a1 = loadA(buf, wm * 32 + 16);

        // Software-pipelined B fragments: load tile j+1 before consuming j.
        v16bf b_cur = loadB(buf, wn * 64);
#pragma unroll
        for (int j = 0; j < 4; ++j) {
            v16bf b_nxt = (j < 3) ? loadB(buf, wn * 64 + (j + 1) * 16) : b_cur;
            acc[0][j] = __builtin_amdgcn_wmma_f32_16x16x32_bf16(
                false, a0, false, b_cur, (short)0, acc[0][j], false, false);
            acc[1][j] = __builtin_amdgcn_wmma_f32_16x16x32_bf16(
                false, a1, false, b_cur, (short)0, acc[1][j], false, false);
            b_cur = b_nxt;
        }

        // Drain this wave's async tile loads, then barrier: after the
        // barrier every wave's buf^1 data is visible in LDS.
        asm volatile("s_wait_asynccnt 0x0" ::: "memory");
        __syncthreads();
    }

    // ---- epilogue: f32 C/D layout: lane -> N, VGPR v -> M = lh*8 + v -----
#pragma unroll
    for (int i = 0; i < 2; ++i) {
        const int mrow = bm + wm * 32 + i * 16 + lh * 8;
#pragma unroll
        for (int j = 0; j < 4; ++j) {
            const int col = bn + wn * 64 + j * 16 + lr;
            const float bv = bias[col];
#pragma unroll
            for (int v = 0; v < 8; ++v) {
                C[(long long)(mrow + v) * N + col] = acc[i][j][v] + bv;
            }
        }
    }
}

// ---------------------------------------------------------------------------
// Host launcher
// ---------------------------------------------------------------------------
extern "C" void kernel_launch(void* const* d_in, const int* in_sizes, int n_in,
                              void* d_out, int out_size, void* d_ws, size_t ws_size,
                              hipStream_t stream) {
    (void)n_in; (void)out_size;
    const float* x      = (const float*)d_in[0];   // [B,S,K] fp32
    const int*   codes  = (const int*)d_in[1];     // [N] int32
    const float* basis  = (const float*)d_in[2];   // [256,K] fp32
    const int*   rq     = (const int*)d_in[3];     // [N,K] int32
    const float* scales = (const float*)d_in[4];   // [N] fp32
    const float* bias   = (const float*)d_in[5];   // [N] fp32
    float* out = (float*)d_out;

    const int N = in_sizes[1];                 // 4096
    const int K = in_sizes[2] / 256;           // 4096 (basis is [256,K])
    const int M = (int)(in_sizes[0] / K);      // 8192 (B*S)

    const size_t needW = (size_t)N * K * sizeof(__bf16);   // 33.5 MB
    const size_t needX = (size_t)M * K * sizeof(__bf16);   // 67 MB

    __bf16* Wb = (__bf16*)d_ws;

    {   // decode weights -> bf16
        long long threads = (long long)N * K / 8;
        int blocks = (int)((threads + 255) / 256);
        decode_w_kernel<<<blocks, 256, 0, stream>>>(codes, basis, rq, scales, Wb, K);
    }

    dim3 grid(N / BN, M / BM);
    if (ws_size >= needW + needX) {
        // Preferred path: pre-convert x, stage both tiles with async-to-LDS.
        __bf16* Xb = (__bf16*)((char*)d_ws + needW);
        long long threads = (long long)M * K / 8;
        int blocks = (int)((threads + 255) / 256);
        cvt_x_kernel<<<blocks, 256, 0, stream>>>(x, Xb);
        gemm_bf16_kernel<true><<<grid, 256, 0, stream>>>(x, Xb, Wb, bias, out, M, N, K);
    } else {
        // Small-workspace fallback: convert x during staging.
        gemm_bf16_kernel<false><<<grid, 256, 0, stream>>>(x, nullptr, Wb, bias, out, M, N, K);
    }
}
